// Informer_56891136803472
// MI455X (gfx1250) — compile-verified, matrix operand order checked by reading, not run on hardware
//
#include <hip/hip_runtime.h>
#include <math.h>

// ---------------------------------------------------------------------------
// Informer forward for MI455X (gfx1250, wave32, WMMA).
// All dense contractions go through v_wmma_f32_16x16x32_bf16.
// B tiles are staged in LDS (shared by all 8 waves of a block); the fill uses
// the CDNA5 async global->LDS path (ASYNCcnt) when the builtin is available,
// double-buffered so the next tile's copy overlaps the current tile's WMMAs.
// ---------------------------------------------------------------------------

typedef __bf16 bf16_t;
typedef __attribute__((ext_vector_type(16))) __bf16 v16bf;
typedef __attribute__((ext_vector_type(8)))  __bf16 v8bf;
typedef __attribute__((ext_vector_type(8)))  float  v8f;
typedef int v4i __attribute__((vector_size(16)));   // builtin's b128 unit

#define Bc    4
#define CTXc  1024
#define PREDc 512
#define PADc  512
#define CINc  7
#define TFc   4
#define Dc    512
#define Hc    8
#define FFc   512
#define DKc   64

#if defined(__has_builtin)
#if __has_builtin(__builtin_amdgcn_global_load_async_to_lds_b128) && \
    __has_builtin(__builtin_amdgcn_s_wait_asynccnt)
#define HAVE_ASYNC_LDS 1
#endif
#endif

#define GLOBAL_AS __attribute__((address_space(1)))
#define LDS_AS    __attribute__((address_space(3)))

__device__ __forceinline__ float gelu_f(float x) {
    return 0.5f * x * (1.f + erff(x * 0.7071067811865475f));
}

// Load one 16-bit WMMA fragment half-row: elements [0..7] and [16..23]
// relative to p (p already includes the per-lane half*8 K offset).
// All strides are multiples of 8 elements -> 16B-aligned vector loads.
__device__ __forceinline__ v16bf load_frag(const bf16_t* p) {
    v8bf lo = *reinterpret_cast<const v8bf*>(p);
    v8bf hi = *reinterpret_cast<const v8bf*>(p + 16);
    return __builtin_shufflevector(lo, hi, 0,1,2,3,4,5,6,7,8,9,10,11,12,13,14,15);
}

// LDS B-tile geometry: 64 rows (N) x 32 (K) bf16, row stride padded to 40
// elements (80 B) -> 16B-aligned ds_load_b128 at minimum 2-way bank access.
#define BT_STRIDE 40
#define BT_ELEMS  (64 * BT_STRIDE)

// ---------------------------------------------------------------------------
// Batched GEMM: C[z] = act(scale * A[z] (M,K) x Bt[z]^T (N,K) + bias)
// One wave computes a 16(M) x 64(N) strip: A-fragment reused over 4 WMMAs.
// 8 waves per block -> 128 x 64 tile.  K must be a multiple of 32.
// ---------------------------------------------------------------------------
__global__ __launch_bounds__(256)
void wmma_gemm(const bf16_t* __restrict__ A,  int lda, long long sAb,
               const bf16_t* __restrict__ Bt, int ldb, long long sBb,
               float* __restrict__ Cf, bf16_t* __restrict__ Cb,
               int ldc, long long sCb,
               const float* __restrict__ bias,
               int M, int N, int K, float scale, int act)
{
    __shared__ bf16_t bsm[2][BT_ELEMS];

    const int tid  = threadIdx.x;
    const int wave = tid >> 5;
    const int lane = tid & 31;
    const int hlf  = lane >> 4;      // which K-half this lane holds
    const int l16  = lane & 15;
    const int zb   = blockIdx.z;

    A  += (long long)zb * sAb;
    Bt += (long long)zb * sBb;

    const int m0 = blockIdx.x * 128 + wave * 16;
    const int n0 = blockIdx.y * 64;

    int arow = m0 + l16; if (arow >= M) arow = M - 1;   // clamp; guarded at store
    const bf16_t* Ap = A + (long long)arow * lda + hlf * 8;

    // Cooperative B-tile staging: 256 threads x one 16B chunk = 64 rows x 64B.
    const int brow = tid >> 2;          // 0..63 : N row within tile
    const int bseg = tid & 3;           // which 16B chunk of the 32-K row
    int bn = n0 + brow; if (bn >= N) bn = N - 1;
    const bf16_t* Bg  = Bt + (long long)bn * ldb + bseg * 8;
    const int     lofs = brow * BT_STRIDE + bseg * 8;   // element offset in buffer

    v8f acc[4] = {};
    const int nk = K >> 5;

#if HAVE_ASYNC_LDS
    // prologue: async-fill tile 0 into buffer 0
    __builtin_amdgcn_global_load_async_to_lds_b128(
        (GLOBAL_AS v4i*)(Bg), (LDS_AS v4i*)(&bsm[0][lofs]), 0, 0);
    for (int i = 0; i < nk; ++i) {
        __builtin_amdgcn_s_wait_asynccnt(0);   // tile i resident (this wave's chunks)
        __syncthreads();                       // ...and everyone else's
        if (i + 1 < nk)                        // overlap next fill with compute
            __builtin_amdgcn_global_load_async_to_lds_b128(
                (GLOBAL_AS v4i*)(Bg + (i + 1) * 32),
                (LDS_AS v4i*)(&bsm[(i + 1) & 1][lofs]), 0, 0);
        const bf16_t* bb = &bsm[i & 1][0];
        v16bf a = load_frag(Ap + i * 32);
#pragma unroll
        for (int t = 0; t < 4; ++t) {
            v16bf bfrag = load_frag(bb + (t * 16 + l16) * BT_STRIDE + hlf * 8);
            acc[t] = __builtin_amdgcn_wmma_f32_16x16x32_bf16(
                false, a, false, bfrag, (short)0, acc[t], false, false);
        }
    }
#else
    for (int i = 0; i < nk; ++i) {
        v8bf c0 = *reinterpret_cast<const v8bf*>(Bg + i * 32);  // 16B chunk
        __syncthreads();                       // previous tile fully consumed
        *reinterpret_cast<v8bf*>(&bsm[0][lofs]) = c0;
        __syncthreads();                       // tile visible to all waves
        const bf16_t* bb = &bsm[0][0];
        v16bf a = load_frag(Ap + i * 32);
#pragma unroll
        for (int t = 0; t < 4; ++t) {
            v16bf bfrag = load_frag(bb + (t * 16 + l16) * BT_STRIDE + hlf * 8);
            acc[t] = __builtin_amdgcn_wmma_f32_16x16x32_bf16(
                false, a, false, bfrag, (short)0, acc[t], false, false);
        }
    }
#endif

#pragma unroll
    for (int t = 0; t < 4; ++t) {
        int ccol = n0 + t * 16 + l16;
        if (ccol < N) {
            float bv = bias ? bias[ccol] : 0.f;
#pragma unroll
            for (int i = 0; i < 8; ++i) {
                int r = m0 + hlf * 8 + i;          // C layout: lanes16-31 hold M+8
                if (r < M) {
                    float val = acc[t][i] * scale + bv;
                    if (act == 1) val = gelu_f(val);
                    long long off = (long long)zb * sCb + (long long)r * ldc + ccol;
                    if (Cf) Cf[off] = val;
                    if (Cb) Cb[off] = (bf16_t)val;
                }
            }
        }
    }
}

// ---------------------------------------------------------------------------
// Row softmax (optionally causal), f32 in -> bf16 probs out (zero-padded to SP)
// ---------------------------------------------------------------------------
__global__ __launch_bounds__(256)
void softmax_bf16(const float* __restrict__ S, bf16_t* __restrict__ P,
                  int Skv, int SP, int L, int causal)
{
    __shared__ float red[256];
    int r = blockIdx.x;
    int limit = causal ? ((r % L) + 1) : Skv;
    if (limit > Skv) limit = Skv;
    const float* src = S + (long long)r * SP;
    bf16_t*      dst = P + (long long)r * SP;
    int t = threadIdx.x;

    float mx = -3.4e38f;
    for (int j = t; j < limit; j += 256) mx = fmaxf(mx, src[j]);
    red[t] = mx; __syncthreads();
    for (int k = 128; k > 0; k >>= 1) { if (t < k) red[t] = fmaxf(red[t], red[t + k]); __syncthreads(); }
    mx = red[0]; __syncthreads();

    float s = 0.f;
    for (int j = t; j < limit; j += 256) s += __expf(src[j] - mx);
    red[t] = s; __syncthreads();
    for (int k = 128; k > 0; k >>= 1) { if (t < k) red[t] += red[t + k]; __syncthreads(); }
    float inv = 1.f / red[0];

    for (int j = t; j < SP; j += 256)
        dst[j] = (bf16_t)((j < limit) ? __expf(src[j] - mx) * inv : 0.f);
}

// ---------------------------------------------------------------------------
// Fused residual-add + layernorm, writes f32 master + optional bf16 copy.
// One block per row (D=512), 2 elements per thread.
// ---------------------------------------------------------------------------
__global__ __launch_bounds__(256)
void add_layernorm(const float* __restrict__ x, const float* __restrict__ y,
                   const float* __restrict__ g, const float* __restrict__ b,
                   float* __restrict__ outf, bf16_t* __restrict__ outb)
{
    __shared__ float red[256];
    int r = blockIdx.x, t = threadIdx.x;
    long long base = (long long)r * Dc;
    float a0 = x[base + t], a1 = x[base + t + 256];
    if (y) { a0 += y[base + t]; a1 += y[base + t + 256]; }

    red[t] = a0 + a1; __syncthreads();
    for (int k = 128; k > 0; k >>= 1) { if (t < k) red[t] += red[t + k]; __syncthreads(); }
    float mu = red[0] * (1.f / Dc); __syncthreads();

    float d0 = a0 - mu, d1 = a1 - mu;
    red[t] = d0 * d0 + d1 * d1; __syncthreads();
    for (int k = 128; k > 0; k >>= 1) { if (t < k) red[t] += red[t + k]; __syncthreads(); }
    float inv = rsqrtf(red[0] * (1.f / Dc) + 1e-5f);

    float o0 = d0 * inv * g[t] + b[t];
    float o1 = d1 * inv * g[t + 256] + b[t + 256];
    if (outf) { outf[base + t] = o0; outf[base + t + 256] = o1; }
    if (outb) { outb[base + t] = (bf16_t)o0; outb[base + t + 256] = (bf16_t)o1; }
}

// V [B,S,H*DK] bf16 -> vT [B,H,DK,SP] bf16 (zero-padded along S)
__global__ void pack_vT(const bf16_t* __restrict__ V, bf16_t* __restrict__ vT,
                        int Skv, int SP)
{
    long long i = (long long)blockIdx.x * 256 + threadIdx.x;
    long long total = (long long)Bc * Hc * DKc * SP;
    if (i >= total) return;
    int s = (int)(i % SP); long long r = i / SP;
    int d = (int)(r % DKc); r /= DKc;
    int h = (int)(r % Hc);
    int b = (int)(r / Hc);
    bf16_t val = (bf16_t)0.f;
    if (s < Skv) val = V[((long long)b * Skv + s) * Dc + h * DKc + d];
    vT[i] = val;
}

__global__ void cvt_bf16(const float* __restrict__ x, bf16_t* __restrict__ y, long long n)
{
    long long i = (long long)blockIdx.x * 256 + threadIdx.x;
    if (i < n) y[i] = (bf16_t)x[i];
}

// Wt[n*K + k] = (bf16)W[k*N + n]
__global__ void transpose_bf16(const float* __restrict__ W, bf16_t* __restrict__ Wt,
                               int K, int N)
{
    long long i = (long long)blockIdx.x * 256 + threadIdx.x;
    if (i >= (long long)K * N) return;
    int n = (int)(i / K), k = (int)(i % K);
    Wt[i] = (bf16_t)W[(long long)k * N + n];
}

// circular-pad(2) im2col for conv_distil: col[(b*Lc+t)*1536 + c*3+j]
__global__ void im2col_circ(const float* __restrict__ x, bf16_t* __restrict__ col,
                            int L, int Lc)
{
    long long i = (long long)blockIdx.x * 256 + threadIdx.x;
    long long total = (long long)Bc * Lc * Dc * 3;
    if (i >= total) return;
    int j = (int)(i % 3); long long r = i / 3;
    int c = (int)(r % Dc); long long q = r / Dc;
    int t = (int)(q % Lc);
    int b = (int)(q / Lc);
    int ti = t + j - 2; ti %= L; if (ti < 0) ti += L;
    col[i] = (bf16_t)x[((long long)b * L + ti) * Dc + c];
}

// per-channel batchnorm statistics over all (b,t)
__global__ __launch_bounds__(256)
void bn_stats(const float* __restrict__ y, float* __restrict__ st, int rows)
{
    __shared__ float rs[256], rq[256];
    int d = blockIdx.x, t = threadIdx.x;
    float s = 0.f, q = 0.f;
    for (int r = t; r < rows; r += 256) {
        float v = y[(long long)r * Dc + d]; s += v; q += v * v;
    }
    rs[t] = s; rq[t] = q; __syncthreads();
    for (int k = 128; k > 0; k >>= 1) {
        if (t < k) { rs[t] += rs[t + k]; rq[t] += rq[t + k]; } __syncthreads();
    }
    if (t == 0) { float mu = rs[0] / rows; st[d] = mu; st[Dc + d] = rq[0] / rows - mu * mu; }
}

// fused BN-apply + ELU + (-inf pad, window3 stride2) maxpool
__global__ void bn_elu_pool(const float* __restrict__ y, const float* __restrict__ st,
                            const float* __restrict__ g, const float* __restrict__ b,
                            float* __restrict__ out, int Lc, int Lo)
{
    long long i = (long long)blockIdx.x * 256 + threadIdx.x;
    long long total = (long long)Bc * Lo * Dc;
    if (i >= total) return;
    int d = (int)(i % Dc); long long r = i / Dc;
    int t = (int)(r % Lo);
    int bb = (int)(r / Lo);
    float sc = g[d] * rsqrtf(st[Dc + d] + 1e-5f);
    float bs = b[d] - st[d] * sc;
    float best = -3.4e38f;
    for (int w = 0; w < 3; ++w) {
        int p = 2 * t + w;                 // padded coords [0, Lc+1]
        if (p >= 1 && p <= Lc) {
            float v = y[((long long)bb * Lc + (p - 1)) * Dc + d] * sc + bs;
            v = v > 0.f ? v : (__expf(v) - 1.f);
            best = fmaxf(best, v);
        }
    }
    out[i] = best;
}

// data_embedding: circular tok-conv + positional + time-feature linear.
// mode 0 = encoder (x = past_target, mark = past_time)
// mode 1 = decoder (x = concat(past[-512:], 0), mark = concat(past_t[-512:], fut_t))
__global__ void embed_kernel(const float* __restrict__ pt, const float* __restrict__ ptf,
                             const float* __restrict__ ftf,
                             const float* __restrict__ tokw,
                             const float* __restrict__ timew, const float* __restrict__ timeb,
                             float* __restrict__ out, int L, int mode)
{
    long long i = (long long)blockIdx.x * 256 + threadIdx.x;
    long long total = (long long)Bc * L * Dc;
    if (i >= total) return;
    int d = (int)(i % Dc); long long r = i / Dc;
    int t = (int)(r % L);
    int b = (int)(r / L);

    float acc = timeb[d];
    for (int f = 0; f < TFc; ++f) {
        float mv;
        if (mode == 0) mv = ptf[((long long)b * CTXc + t) * TFc + f];
        else mv = (t < PADc) ? ptf[((long long)b * CTXc + (CTXc - PADc) + t) * TFc + f]
                             : ftf[((long long)b * PREDc + (t - PADc)) * TFc + f];
        acc += mv * timew[f * Dc + d];
    }
    for (int j = 0; j < 3; ++j) {
        int ti = t + j - 1; ti = (ti % L + L) % L;
        for (int c = 0; c < CINc; ++c) {
            float xv;
            if (mode == 0) xv = pt[((long long)b * CTXc + ti) * CINc + c];
            else xv = (ti < PADc) ? pt[((long long)b * CTXc + (CTXc - PADc) + ti) * CINc + c] : 0.f;
            acc += xv * tokw[(d * CINc + c) * 3 + j];
        }
    }
    int i2 = d & ~1;
    float div = __expf(-(float)i2 * (9.2103403719761836f / (float)Dc));
    float ang = (float)t * div;
    acc += (d & 1) ? __cosf(ang) : __sinf(ang);
    out[i] = acc;
}

// final projection D -> C_IN, only last PRED positions
__global__ void proj_out_k(const float* __restrict__ x, const float* __restrict__ pw,
                           const float* __restrict__ pb, float* __restrict__ out)
{
    long long i = (long long)blockIdx.x * 256 + threadIdx.x;
    long long total = (long long)Bc * PREDc * CINc;
    if (i >= total) return;
    int c = (int)(i % CINc); long long r = i / CINc;
    int t = (int)(r % PREDc);
    int b = (int)(r / PREDc);
    const float* row = x + ((long long)b * (PADc + PREDc) + PADc + t) * Dc;
    float acc = pb[c];
    for (int d = 0; d < Dc; ++d) acc += row[d] * pw[d * CINc + c];
    out[i] = acc;
}

// ---------------------------------------------------------------------------
// Host orchestration
// ---------------------------------------------------------------------------
extern "C" void kernel_launch(void* const* d_in, const int* in_sizes, int n_in,
                              void* d_out, int out_size, void* d_ws, size_t ws_size,
                              hipStream_t stream)
{
    (void)in_sizes; (void)n_in; (void)out_size; (void)ws_size;
    auto F = [&](int i) { return (const float*)d_in[i]; };

    char* wsb = (char*)d_ws; size_t off = 0;
    auto alloc = [&](size_t bytes) -> void* {
        void* r = wsb + off; off = (off + bytes + 255) & ~(size_t)255; return r;
    };

    // ---- unpack inputs (setup_inputs dict order, nested dicts in insertion order)
    const float* past_target = F(0);
    const float* past_time   = F(1);
    const float* future_time = F(2);
    int ip = 3;
    const float* enc_tok = F(ip++);
    const float* enc_tw  = F(ip++); const float* enc_tb = F(ip++);
    const float* dec_tok = F(ip++);
    const float* dec_tw  = F(ip++); const float* dec_tb = F(ip++);

    struct AttnW { const float *qw,*qb,*kw,*kb,*vw,*vb,*ow,*ob; };
    struct EncL  { AttnW a; const float *f1w,*f1b,*f2w,*f2b,*n1g,*n1b,*n2g,*n2b; };
    struct ConvL { const float *w,*b,*g,*bb; };
    struct DecL  { AttnW s,c; const float *f1w,*f1b,*f2w,*f2b,*n1g,*n1b,*n2g,*n2b,*n3g,*n3b; };

    auto rdA = [&]() { AttnW a; a.qw=F(ip++);a.qb=F(ip++);a.kw=F(ip++);a.kb=F(ip++);
                       a.vw=F(ip++);a.vb=F(ip++);a.ow=F(ip++);a.ob=F(ip++); return a; };
    EncL enc[3];
    for (int i=0;i<3;++i){ enc[i].a=rdA();
        enc[i].f1w=F(ip++);enc[i].f1b=F(ip++);enc[i].f2w=F(ip++);enc[i].f2b=F(ip++);
        enc[i].n1g=F(ip++);enc[i].n1b=F(ip++);enc[i].n2g=F(ip++);enc[i].n2b=F(ip++); }
    ConvL cv[2];
    for (int i=0;i<2;++i){ cv[i].w=F(ip++);cv[i].b=F(ip++);cv[i].g=F(ip++);cv[i].bb=F(ip++); }
    const float* encN_g=F(ip++); const float* encN_b=F(ip++);
    DecL dcl[2];
    for (int i=0;i<2;++i){ dcl[i].s=rdA(); dcl[i].c=rdA();
        dcl[i].f1w=F(ip++);dcl[i].f1b=F(ip++);dcl[i].f2w=F(ip++);dcl[i].f2b=F(ip++);
        dcl[i].n1g=F(ip++);dcl[i].n1b=F(ip++);dcl[i].n2g=F(ip++);dcl[i].n2b=F(ip++);
        dcl[i].n3g=F(ip++);dcl[i].n3b=F(ip++); }
    const float* decN_g=F(ip++); const float* decN_b=F(ip++);
    const float* proj_w=F(ip++); const float* proj_b=F(ip++);

    // ---- weight prep: transpose+convert to bf16 [N,K] so WMMA B-fragments
    //      are contiguous aligned 16B loads
    auto prepT = [&](const float* W, int K, int N) -> bf16_t* {
        bf16_t* wt = (bf16_t*)alloc((size_t)K * N * sizeof(bf16_t));
        long long n = (long long)K * N;
        transpose_bf16<<<(unsigned)((n + 255) / 256), 256, 0, stream>>>(W, wt, K, N);
        return wt;
    };
    auto prepC = [&](const float* W, long long n) -> bf16_t* {   // already [N,K]
        bf16_t* wt = (bf16_t*)alloc((size_t)n * sizeof(bf16_t));
        cvt_bf16<<<(unsigned)((n + 255) / 256), 256, 0, stream>>>(W, wt, n);
        return wt;
    };

    bf16_t *EQ[3],*EK[3],*EV[3],*EO[3],*EF1[3],*EF2[3];
    for (int i=0;i<3;++i){
        EQ[i]=prepT(enc[i].a.qw,Dc,Dc); EK[i]=prepT(enc[i].a.kw,Dc,Dc);
        EV[i]=prepT(enc[i].a.vw,Dc,Dc); EO[i]=prepT(enc[i].a.ow,Dc,Dc);
        EF1[i]=prepT(enc[i].f1w,Dc,FFc); EF2[i]=prepT(enc[i].f2w,FFc,Dc);
    }
    bf16_t* CW[2];
    for (int i=0;i<2;++i) CW[i]=prepC(cv[i].w, (long long)Dc*Dc*3);
    bf16_t *SQ[2],*SK[2],*SV[2],*SO[2],*CQ[2],*CK[2],*CVw[2],*CO[2],*DF1[2],*DF2[2];
    for (int j=0;j<2;++j){
        SQ[j]=prepT(dcl[j].s.qw,Dc,Dc); SK[j]=prepT(dcl[j].s.kw,Dc,Dc);
        SV[j]=prepT(dcl[j].s.vw,Dc,Dc); SO[j]=prepT(dcl[j].s.ow,Dc,Dc);
        CQ[j]=prepT(dcl[j].c.qw,Dc,Dc); CK[j]=prepT(dcl[j].c.kw,Dc,Dc);
        CVw[j]=prepT(dcl[j].c.vw,Dc,Dc); CO[j]=prepT(dcl[j].c.ow,Dc,Dc);
        DF1[j]=prepT(dcl[j].f1w,Dc,FFc); DF2[j]=prepT(dcl[j].f2w,FFc,Dc);
    }

    // ---- activation buffers
    const int MAXR = Bc * CTXc;                        // 4096 rows
    float*  xe_f = (float*) alloc((size_t)MAXR*Dc*4);
    bf16_t* xe_b = (bf16_t*)alloc((size_t)MAXR*Dc*2);
    float*  xd_f = (float*) alloc((size_t)MAXR*Dc*4);
    bf16_t* xd_b = (bf16_t*)alloc((size_t)MAXR*Dc*2);
    float*  t1   = (float*) alloc((size_t)MAXR*Dc*4);
    bf16_t* q_b  = (bf16_t*)alloc((size_t)MAXR*Dc*2);
    bf16_t* k_b  = (bf16_t*)alloc((size_t)MAXR*Dc*2);
    bf16_t* v_b  = (bf16_t*)alloc((size_t)MAXR*Dc*2);
    bf16_t* ctx_b= (bf16_t*)alloc((size_t)MAXR*Dc*2);
    bf16_t* ff_b = (bf16_t*)alloc((size_t)MAXR*FFc*2);
    bf16_t* vT   = (bf16_t*)alloc((size_t)Bc*Hc*DKc*1024*2);
    float*  scor = (float*) alloc((size_t)Bc*1024*1024*4);
    bf16_t* prob = (bf16_t*)alloc((size_t)Bc*1024*1024*2);
    bf16_t* colb = (bf16_t*)alloc((size_t)Bc*(CTXc+2)*3*Dc*2);
    float*  ycv  = (float*) alloc((size_t)Bc*(CTXc+2)*Dc*4);
    float*  bst  = (float*) alloc((size_t)2*Dc*4);

    auto gemm = [&](const bf16_t* A, int lda, long long sAb,
                    const bf16_t* Bt, int ldb, long long sBb,
                    float* Cf, bf16_t* Cb, int ldc, long long sCb,
                    const float* bias, int M, int N, int K, int batch,
                    float scale, int act) {
        dim3 g((M + 127) / 128, (N + 63) / 64, batch);
        wmma_gemm<<<g, dim3(256), 0, stream>>>(A, lda, sAb, Bt, ldb, sBb,
                                               Cf, Cb, ldc, sCb, bias, M, N, K, scale, act);
    };

    // full multi-head attention, per-head host loop (caps workspace)
    auto attention = [&](const bf16_t* q, const bf16_t* k, const bf16_t* v,
                         int Lq, int Skv, int causal, int mix, bf16_t* ctx) {
        int SP = (Skv + 31) & ~31;
        long long tot = (long long)Bc * Hc * DKc * SP;
        pack_vT<<<(unsigned)((tot + 255) / 256), 256, 0, stream>>>(v, vT, Skv, SP);
        for (int h = 0; h < Hc; ++h) {
            // scores = Q K^T / sqrt(DK)   (M=Lq, N=Skv, K=64)
            gemm(q + h * DKc, Dc, (long long)Lq * Dc,
                 k + h * DKc, Dc, (long long)Skv * Dc,
                 scor, nullptr, SP, (long long)Lq * SP,
                 nullptr, Lq, Skv, DKc, Bc, 0.125f, 0);
            softmax_bf16<<<Bc * Lq, 256, 0, stream>>>(scor, prob, Skv, SP, Lq, causal);
            // ctx = P V   (M=Lq, N=64, K=SP; zero-padded tail contributes 0)
            long long sCh = mix ? (long long)Lq * DKc : (long long)DKc;
            int ldcc = mix ? DKc : Dc;
            gemm(prob, SP, (long long)Lq * SP,
                 vT + (long long)h * DKc * SP, SP, (long long)Hc * DKc * SP,
                 nullptr, ctx + h * sCh, ldcc, (long long)Lq * Dc,
                 nullptr, Lq, DKc, SP, Bc, 1.f, 0);
        }
    };

    // ================= ENCODER =================
    {
        long long n = (long long)Bc * CTXc * Dc;
        embed_kernel<<<(unsigned)((n + 255) / 256), 256, 0, stream>>>(
            past_target, past_time, future_time, enc_tok, enc_tw, enc_tb, xe_f, CTXc, 0);
    }
    int L = CTXc;
    for (int i = 0; i < 3; ++i) {
        int rows = Bc * L;
        long long n = (long long)rows * Dc;
        cvt_bf16<<<(unsigned)((n + 255) / 256), 256, 0, stream>>>(xe_f, xe_b, n);
        gemm(xe_b, Dc, 0, EQ[i], Dc, 0, nullptr, q_b, Dc, 0, enc[i].a.qb, rows, Dc, Dc, 1, 1.f, 0);
        gemm(xe_b, Dc, 0, EK[i], Dc, 0, nullptr, k_b, Dc, 0, enc[i].a.kb, rows, Dc, Dc, 1, 1.f, 0);
        gemm(xe_b, Dc, 0, EV[i], Dc, 0, nullptr, v_b, Dc, 0, enc[i].a.vb, rows, Dc, Dc, 1, 1.f, 0);
        attention(q_b, k_b, v_b, L, L, 0, 0, ctx_b);
        gemm(ctx_b, Dc, 0, EO[i], Dc, 0, t1, nullptr, Dc, 0, enc[i].a.ob, rows, Dc, Dc, 1, 1.f, 0);
        add_layernorm<<<rows, 256, 0, stream>>>(xe_f, t1, enc[i].n1g, enc[i].n1b, xe_f, xe_b);
        gemm(xe_b, Dc, 0, EF1[i], Dc, 0, nullptr, ff_b, FFc, 0, enc[i].f1b, rows, FFc, Dc, 1, 1.f, 1);
        gemm(ff_b, FFc, 0, EF2[i], FFc, 0, t1, nullptr, Dc, 0, enc[i].f2b, rows, Dc, FFc, 1, 1.f, 0);
        add_layernorm<<<rows, 256, 0, stream>>>(xe_f, t1, enc[i].n2g, enc[i].n2b, xe_f, xe_b);
        if (i < 2) {   // conv distil
            int Lc2 = L + 2, Lo = (L + 1) / 2 + 1;
            long long nc = (long long)Bc * Lc2 * 3 * Dc;
            im2col_circ<<<(unsigned)((nc + 255) / 256), 256, 0, stream>>>(xe_f, colb, L, Lc2);
            gemm(colb, 3 * Dc, 0, CW[i], 3 * Dc, 0, ycv, nullptr, Dc, 0,
                 cv[i].b, Bc * Lc2, Dc, 3 * Dc, 1, 1.f, 0);
            bn_stats<<<Dc, 256, 0, stream>>>(ycv, bst, Bc * Lc2);
            long long npo = (long long)Bc * Lo * Dc;
            bn_elu_pool<<<(unsigned)((npo + 255) / 256), 256, 0, stream>>>(
                ycv, bst, cv[i].g, cv[i].bb, xe_f, Lc2, Lo);
            L = Lo;
        }
    }
    int Lenc = L;
    add_layernorm<<<Bc * Lenc, 256, 0, stream>>>(xe_f, nullptr, encN_g, encN_b, xe_f, xe_b);

    // ================= DECODER =================
    const int Ld = PADc + PREDc;   // 1024
    {
        long long n = (long long)Bc * Ld * Dc;
        embed_kernel<<<(unsigned)((n + 255) / 256), 256, 0, stream>>>(
            past_target, past_time, future_time, dec_tok, dec_tw, dec_tb, xd_f, Ld, 1);
    }
    for (int j = 0; j < 2; ++j) {
        int rows = Bc * Ld;
        long long n = (long long)rows * Dc;
        cvt_bf16<<<(unsigned)((n + 255) / 256), 256, 0, stream>>>(xd_f, xd_b, n);
        // self attention (causal, mix=True)
        gemm(xd_b, Dc, 0, SQ[j], Dc, 0, nullptr, q_b, Dc, 0, dcl[j].s.qb, rows, Dc, Dc, 1, 1.f, 0);
        gemm(xd_b, Dc, 0, SK[j], Dc, 0, nullptr, k_b, Dc, 0, dcl[j].s.kb, rows, Dc, Dc, 1, 1.f, 0);
        gemm(xd_b, Dc, 0, SV[j], Dc, 0, nullptr, v_b, Dc, 0, dcl[j].s.vb, rows, Dc, Dc, 1, 1.f, 0);
        attention(q_b, k_b, v_b, Ld, Ld, 1, 1, ctx_b);
        gemm(ctx_b, Dc, 0, SO[j], Dc, 0, t1, nullptr, Dc, 0, dcl[j].s.ob, rows, Dc, Dc, 1, 1.f, 0);
        add_layernorm<<<rows, 256, 0, stream>>>(xd_f, t1, dcl[j].n1g, dcl[j].n1b, xd_f, xd_b);
        // cross attention vs encoder output
        int rowsE = Bc * Lenc;
        gemm(xd_b, Dc, 0, CQ[j], Dc, 0, nullptr, q_b, Dc, 0, dcl[j].c.qb, rows, Dc, Dc, 1, 1.f, 0);
        gemm(xe_b, Dc, 0, CK[j], Dc, 0, nullptr, k_b, Dc, 0, dcl[j].c.kb, rowsE, Dc, Dc, 1, 1.f, 0);
        gemm(xe_b, Dc, 0, CVw[j], Dc, 0, nullptr, v_b, Dc, 0, dcl[j].c.vb, rowsE, Dc, Dc, 1, 1.f, 0);
        attention(q_b, k_b, v_b, Ld, Lenc, 0, 0, ctx_b);
        gemm(ctx_b, Dc, 0, CO[j], Dc, 0, t1, nullptr, Dc, 0, dcl[j].c.ob, rows, Dc, Dc, 1, 1.f, 0);
        add_layernorm<<<rows, 256, 0, stream>>>(xd_f, t1, dcl[j].n2g, dcl[j].n2b, xd_f, xd_b);
        // feed-forward
        gemm(xd_b, Dc, 0, DF1[j], Dc, 0, nullptr, ff_b, FFc, 0, dcl[j].f1b, rows, FFc, Dc, 1, 1.f, 1);
        gemm(ff_b, FFc, 0, DF2[j], FFc, 0, t1, nullptr, Dc, 0, dcl[j].f2b, rows, Dc, FFc, 1, 1.f, 0);
        add_layernorm<<<rows, 256, 0, stream>>>(xd_f, t1, dcl[j].n3g, dcl[j].n3b, xd_f, xd_b);
    }
    add_layernorm<<<Bc * Ld, 256, 0, stream>>>(xd_f, nullptr, decN_g, decN_b, xd_f, nullptr);
    {
        long long np = (long long)Bc * PREDc * CINc;
        proj_out_k<<<(unsigned)((np + 255) / 256), 256, 0, stream>>>(
            xd_f, proj_w, proj_b, (float*)d_out);
    }
}
